// Model_39676907884671
// MI455X (gfx1250) — matrix operand order repaired and checked by reading, not compiled, not run on hardware
//
#include <hip/hip_runtime.h>

// CDNA5 (gfx1250, wave32) fused attention, register-blocked:
//   stage 0: f32 -> bf16 conversion of activations + weights (streaming)
//   stage 1: q/k/v projections, 2Mx4N register tiles per wave on
//            v_wmma_f32_16x16x32_bf16 (v stored transposed as vT[j][h][n])
//   stage 2: per-j logits (2Mx2N tiles) -> masked softmax (LDS) -> PV GEMM
//            (4 p-fragments reused over 32 h-tiles), fused in one kernel.
// All GEMM fragments are two contiguous 16B loads per lane (contiguous k order, same in A and B).

typedef __attribute__((ext_vector_type(16))) __bf16 v16bf;
typedef __attribute__((ext_vector_type(8)))  __bf16 v8bf;
typedef __attribute__((ext_vector_type(4)))  __bf16 v4bf;
typedef __attribute__((ext_vector_type(8)))  float  v8f;
typedef __attribute__((ext_vector_type(4)))  float  v4f;

#define H_DIM 1024
#define S_DIM 256
#define N_HEADS 32

// ---- fragment helpers (contiguous-k layouts, wave32) -------------------------

// A-matrix 16x32 bf16: lane l -> row l%16; elems 0..15 = K (l/16)*16+0..15.
// `rowk` = &A[row][kk].
__device__ inline v16bf load_frag_a(const __bf16* rowk, int lhi) {
    const v8bf lo = *reinterpret_cast<const v8bf*>(rowk + lhi * 16);
    const v8bf hi = *reinterpret_cast<const v8bf*>(rowk + lhi * 16 + 8);
    v16bf r;
#pragma unroll
    for (int e = 0; e < 8; ++e) { r[e] = lo[e]; r[8 + e] = hi[e]; }
    return r;
}

// B-matrix 32x16 bf16: lane l -> col l%16; elems 0..15 = K (l/16)*16+0..15.
// `colk` = &Bcol[kk + (l/16)*16]  (contiguous 16 bf16).
__device__ inline v16bf load_frag_b(const __bf16* colk) {
    const v8bf lo = *reinterpret_cast<const v8bf*>(colk);
    const v8bf hi = *reinterpret_cast<const v8bf*>(colk + 8);
    v16bf r;
#pragma unroll
    for (int e = 0; e < 8; ++e) { r[e] = lo[e]; r[8 + e] = hi[e]; }
    return r;
}

__device__ inline v8f wmma_bf16(v16bf a, v16bf b, v8f c) {
    return __builtin_amdgcn_wmma_f32_16x16x32_bf16(
        false, a, false, b, (short)0, c, false, false);
}

// ---- stage 0: f32 -> bf16 --------------------------------------------------

__global__ __launch_bounds__(256) void cvt_f32_bf16(const float* __restrict__ s,
                                                    __bf16* __restrict__ d, int n) {
    int i = (blockIdx.x * blockDim.x + threadIdx.x) * 4;
    if (i + 4 <= n) {
        v4f x = *reinterpret_cast<const v4f*>(s + i);
        v4bf y;
#pragma unroll
        for (int e = 0; e < 4; ++e) y[e] = (__bf16)x[e];
        *reinterpret_cast<v4bf*>(d + i) = y;
    }
}

// ---- stage 1: projection  Y[M,1024] = X[M,1024] @ W^T + b ------------------
// grid (M/128, 8), block 256 (8 waves). Wave tile: 32 rows x 64 cols
// (2 M-tiles x 4 N-tiles, 8 accumulators; B frags reused across M).
// transpose_v: store Y as vT[j=row/32][h=col][n=row%32] for the PV GEMM.

__global__ __launch_bounds__(256) void proj_kernel(const __bf16* __restrict__ X,
                                                   const __bf16* __restrict__ W,
                                                   const float* __restrict__ bias,
                                                   __bf16* __restrict__ Y,
                                                   int transpose_v) {
    const int lane = threadIdx.x & 31;
    const int wave = threadIdx.x >> 5;
    const int lrow = lane & 15;
    const int lhi  = lane >> 4;
    const int m0 = blockIdx.x * 128 + (wave >> 1) * 32;  // rows m0..m0+31
    const int n0 = blockIdx.y * 128 + (wave & 1) * 64;   // cols n0..n0+63

    const __bf16* arow0 = X + (size_t)(m0 + lrow) * H_DIM;
    const __bf16* arow1 = arow0 + (size_t)16 * H_DIM;
    v8f acc[2][4] = {};

    for (int kk = 0; kk < H_DIM; kk += 32) {
        v16bf a0 = load_frag_a(arow0 + kk, lhi);
        v16bf a1 = load_frag_a(arow1 + kk, lhi);
#pragma unroll
        for (int t = 0; t < 4; ++t) {
            const __bf16* bcol =
                W + (size_t)(n0 + t * 16 + lrow) * H_DIM + kk + lhi * 16;
            v16bf b = load_frag_b(bcol);
            acc[0][t] = wmma_bf16(a0, b, acc[0][t]);
            acc[1][t] = wmma_bf16(a1, b, acc[1][t]);
        }
    }

#pragma unroll
    for (int mi = 0; mi < 2; ++mi) {
#pragma unroll
        for (int t = 0; t < 4; ++t) {
            const int n = n0 + t * 16 + lrow;
            const float bn = bias[n];
            if (!transpose_v) {
#pragma unroll
                for (int r = 0; r < 8; ++r) {
                    const int row = m0 + mi * 16 + lhi * 8 + r;
                    Y[(size_t)row * H_DIM + n] = (__bf16)(acc[mi][t][r] + bn);
                }
            } else {
                // rows rowb..rowb+7 share j and have consecutive n-in-head ids
                const int rowb = m0 + mi * 16 + lhi * 8;
                const int j = rowb >> 5, nnb = rowb & 31;
                v8bf pk;
#pragma unroll
                for (int r = 0; r < 8; ++r) pk[r] = (__bf16)(acc[mi][t][r] + bn);
                *reinterpret_cast<v8bf*>(Y + (size_t)j * (H_DIM * N_HEADS) +
                                         (size_t)n * N_HEADS + nnb) = pk;
            }
        }
    }
}

// ---- stage 2: fused logits -> softmax -> PV --------------------------------
// grid (j=256), block 256 (8 waves); one WG handles all 256 i-rows of one j.
// Stage1: wave w -> i rows w*32..w*32+31, both n-tiles (4 accumulators).
// Softmax: 256 threads, one (i)-row of 32 each, masked, scale 1/32.
// Stage2: wave w -> i rows (w>>1)*64..+63 (4 p-frags), h-half (w&1)*512,
//         32 h-tiles; each B fragment feeds 4 WMMAs.

__global__ __launch_bounds__(256) void attn_kernel(const __bf16* __restrict__ qb,
                                                   const __bf16* __restrict__ kb,
                                                   const __bf16* __restrict__ vtb,
                                                   const unsigned char* __restrict__ mask,
                                                   float* __restrict__ out) {
    __shared__ __align__(16) float  slog[S_DIM * N_HEADS];  // 32 KB
    __shared__ __align__(16) __bf16 sp[S_DIM * N_HEADS];    // 16 KB

    const int j    = blockIdx.x;
    const int lane = threadIdx.x & 31;
    const int wave = threadIdx.x >> 5;
    const int lrow = lane & 15;
    const int lhi  = lane >> 4;

    // ---- logits: a[i, n] = sum_h q[i,h] * k[j,n,h]
    {
        const int i0 = wave * 32;
        const __bf16* arow0 = qb + (size_t)(i0 + lrow) * H_DIM;
        const __bf16* arow1 = arow0 + (size_t)16 * H_DIM;
        const __bf16* brow0 = kb + ((size_t)j * N_HEADS + lrow) * H_DIM;
        const __bf16* brow1 = brow0 + (size_t)16 * H_DIM;
        v8f acc[2][2] = {};
        for (int kk = 0; kk < H_DIM; kk += 32) {
            v16bf a0 = load_frag_a(arow0 + kk, lhi);
            v16bf a1 = load_frag_a(arow1 + kk, lhi);
            v16bf b0 = load_frag_b(brow0 + kk + lhi * 16);
            v16bf b1 = load_frag_b(brow1 + kk + lhi * 16);
            acc[0][0] = wmma_bf16(a0, b0, acc[0][0]);
            acc[0][1] = wmma_bf16(a0, b1, acc[0][1]);
            acc[1][0] = wmma_bf16(a1, b0, acc[1][0]);
            acc[1][1] = wmma_bf16(a1, b1, acc[1][1]);
        }
#pragma unroll
        for (int mi = 0; mi < 2; ++mi)
#pragma unroll
            for (int nt = 0; nt < 2; ++nt)
#pragma unroll
                for (int r = 0; r < 8; ++r)
                    slog[(i0 + mi * 16 + lhi * 8 + r) * N_HEADS + nt * 16 + lrow] =
                        acc[mi][nt][r];
    }
    __syncthreads();

    // ---- masked softmax over n (scale 1/sqrt(1024) = 1/32), mask is (i,n)
    {
        const int row = threadIdx.x;   // == global i
        float x[N_HEADS];
        float mx = -__builtin_inff();
#pragma unroll
        for (int n = 0; n < N_HEADS; ++n) {
            float v = slog[row * N_HEADS + n] * 0.03125f;
            v = mask[row * N_HEADS + n] ? v : -__builtin_inff();
            x[n] = v;
            mx = fmaxf(mx, v);
        }
        float s = 0.f;
#pragma unroll
        for (int n = 0; n < N_HEADS; ++n) { float e = __expf(x[n] - mx); x[n] = e; s += e; }
        const float inv = 1.0f / s;
#pragma unroll
        for (int n = 0; n < N_HEADS; ++n)
            sp[row * N_HEADS + n] = (__bf16)(x[n] * inv);
    }
    __syncthreads();

    // ---- out[j, i, h] = sum_n p[i,n] * v[j,n,h]   (A = p 16x32, one K step)
    {
        const int ig4 = wave >> 1;         // group of 4 i-tiles: rows ig4*64..+63
        const int hh  = wave & 1;          // h half: hh*512..+511
        v16bf ap[4];
#pragma unroll
        for (int q = 0; q < 4; ++q) {
            const int ilocal = ig4 * 64 + q * 16 + lrow;
            const v8bf lo = *reinterpret_cast<const v8bf*>(&sp[ilocal * N_HEADS + lhi * 16]);
            const v8bf hi = *reinterpret_cast<const v8bf*>(&sp[ilocal * N_HEADS + lhi * 16 + 8]);
#pragma unroll
            for (int e = 0; e < 8; ++e) { ap[q][e] = lo[e]; ap[q][8 + e] = hi[e]; }
        }
        const size_t vbase = (size_t)j * (H_DIM * N_HEADS);
        const size_t obase = ((size_t)j * S_DIM + ig4 * 64) * H_DIM;
#pragma unroll 2
        for (int t = 0; t < 32; ++t) {
            const int h0 = hh * 512 + t * 16;
            const __bf16* bcol = vtb + vbase + (size_t)(h0 + lrow) * N_HEADS + lhi * 16;
            v16bf b = load_frag_b(bcol);
#pragma unroll
            for (int q = 0; q < 4; ++q) {
                v8f c = {};
                c = wmma_bf16(ap[q], b, c);
#pragma unroll
                for (int r = 0; r < 8; ++r)
                    out[obase + (size_t)(q * 16 + lhi * 8 + r) * H_DIM + h0 + lrow] = c[r];
            }
        }
    }
}

// ---- host glue -------------------------------------------------------------

extern "C" void kernel_launch(void* const* d_in, const int* in_sizes, int n_in,
                              void* d_out, int out_size, void* d_ws, size_t ws_size,
                              hipStream_t stream) {
    const float* query = (const float*)d_in[0];
    const float* key_x = (const float*)d_in[1];
    const float* value = (const float*)d_in[2];
    const unsigned char* mask = (const unsigned char*)d_in[3];
    const float* Wq = (const float*)d_in[4];
    const float* bq = (const float*)d_in[5];
    const float* Wk = (const float*)d_in[6];
    const float* bk = (const float*)d_in[7];
    const float* Wv = (const float*)d_in[8];
    const float* bv = (const float*)d_in[9];
    float* out = (float*)d_out;

    const size_t QN = (size_t)S_DIM * H_DIM;                 // 262144
    const size_t KN = (size_t)S_DIM * N_HEADS * H_DIM;       // 8388608
    const size_t WN = (size_t)H_DIM * H_DIM;                 // 1048576

    __bf16* ws  = (__bf16*)d_ws;
    __bf16* qx  = ws;            ws += QN;   // bf16 query
    __bf16* kx  = ws;            ws += KN;   // bf16 key_x
    __bf16* vx  = ws;            ws += KN;   // bf16 value
    __bf16* wqb = ws;            ws += WN;
    __bf16* wkb = ws;            ws += WN;
    __bf16* wvb = ws;            ws += WN;
    __bf16* qb  = ws;            ws += QN;   // projected q
    __bf16* kb  = ws;            ws += KN;   // projected k   [j*32+n][h]
    __bf16* vtb = ws;            ws += KN;   // projected v^T [j][h][n]

    auto cvt = [&](const float* s, __bf16* d, size_t n) {
        const int threads = 256, per = 4;
        const int blocks = (int)((n + (size_t)threads * per - 1) / ((size_t)threads * per));
        cvt_f32_bf16<<<blocks, threads, 0, stream>>>(s, d, (int)n);
    };
    cvt(query, qx, QN);
    cvt(key_x, kx, KN);
    cvt(value, vx, KN);
    cvt(Wq, wqb, WN);
    cvt(Wk, wkb, WN);
    cvt(Wv, wvb, WN);

    proj_kernel<<<dim3(S_DIM / 128, 8), 256, 0, stream>>>(qx, wqb, bq, qb, 0);
    proj_kernel<<<dim3(S_DIM * N_HEADS / 128, 8), 256, 0, stream>>>(kx, wkb, bk, kb, 0);
    proj_kernel<<<dim3(S_DIM * N_HEADS / 128, 8), 256, 0, stream>>>(vx, wvb, bv, vtb, 1);

    attn_kernel<<<S_DIM, 256, 0, stream>>>(qb, kb, vtb, mask, out);
}
